// TransformerBlock_120259085226
// MI455X (gfx1250) — matrix operand order repaired and checked by reading, not compiled, not run on hardware
//
#include <hip/hip_runtime.h>

typedef __attribute__((ext_vector_type(16))) __bf16 v16bf;
typedef __attribute__((ext_vector_type(8)))  __bf16 v8bf;
typedef __attribute__((ext_vector_type(8)))  float  v8f;

#define BATCH 8
#define SEQ   2048
#define DIM   512
#define HEADS 8
#define HDIM  64
#define MTOT  (BATCH*SEQ)

static __device__ __forceinline__ v16bf cat16(v8bf lo, v8bf hi) {
  v16bf r;
#pragma unroll
  for (int i = 0; i < 8; ++i) { r[i] = lo[i]; r[i + 8] = hi[i]; }
  return r;
}

// A-fragment (16x32 bf16, MxK) from LDS tile, row-major [16][stride].
static __device__ __forceinline__ v16bf frag_a_lds(const __bf16* tile, int stride, int lane) {
  const int half = lane >> 4, m = lane & 15;
  const __bf16* row = tile + m * stride;
  v8bf p0 = *(const v8bf*)(row + half * 8);
  v8bf p1 = *(const v8bf*)(row + 16 + half * 8);
  return cat16(p0, p1);
}

// B-fragment (32x16 bf16, KxN) from LDS stored transposed: rows = N, cols = K.
static __device__ __forceinline__ v16bf frag_b_lds(const __bf16* tileT, int stride, int lane) {
  const int half = lane >> 4, n = lane & 15;
  const __bf16* row = tileT + n * stride;
  v8bf p0 = *(const v8bf*)(row + half * 8);
  v8bf p1 = *(const v8bf*)(row + 16 + half * 8);
  return cat16(p0, p1);
}

static __device__ __forceinline__ v8bf pack8(const float* f) {
  v8bf t;
#pragma unroll
  for (int i = 0; i < 8; ++i) t[i] = (__bf16)f[i];
  return t;
}

// ---- wave32 16-lane butterfly reductions in pure VALU (v_permlane16_b32) ----
static __device__ __forceinline__ float lane16_perm(float x, int lo, int hi) {
#if __has_builtin(__builtin_amdgcn_permlane16)
  int i = __float_as_int(x);
  i = __builtin_amdgcn_permlane16(i, i, lo, hi, false, false);
  return __int_as_float(i);
#else
  (void)lo; (void)hi;
  return x;
#endif
}
static __device__ __forceinline__ float red16_max(float x) {
#if __has_builtin(__builtin_amdgcn_permlane16)
  x = fmaxf(x, lane16_perm(x, 0x67452301, 0xEFCDAB89)); // xor 1
  x = fmaxf(x, lane16_perm(x, 0x54761032, 0xDCFE98BA)); // xor 2
  x = fmaxf(x, lane16_perm(x, 0x32107654, 0xBA98FEDC)); // xor 4
  x = fmaxf(x, lane16_perm(x, 0xFEDCBA98, 0x76543210)); // xor 8
#else
  for (int off = 1; off < 16; off <<= 1) x = fmaxf(x, __shfl_xor(x, off, 32));
#endif
  return x;
}
static __device__ __forceinline__ float red16_sum(float x) {
#if __has_builtin(__builtin_amdgcn_permlane16)
  x += lane16_perm(x, 0x67452301, 0xEFCDAB89);
  x += lane16_perm(x, 0x54761032, 0xDCFE98BA);
  x += lane16_perm(x, 0x32107654, 0xBA98FEDC);
  x += lane16_perm(x, 0xFEDCBA98, 0x76543210);
#else
  for (int off = 1; off < 16; off <<= 1) x += __shfl_xor(x, off, 32);
#endif
  return x;
}

// ---------------------------------------------------------------------------
// bf16-WMMA GEMM:  C[M,N] = A[M,K] @ B  (+bias)(ReLU)(+Res), f32 or bf16 out.
//   TRANSB==0: B is [K,N] row-major (x @ W)
//   TRANSB==1: B is [N,K] row-major (x @ W.T, torch Linear weight)
// Block tile 128x128, 8 waves (4x2), each wave 32x64 = 2x4 WMMA frags.
// Software-pipelined K loop with double-buffered LDS (one barrier per step).
// ---------------------------------------------------------------------------
template <int TRANSB>
__global__ __launch_bounds__(256) void gemm_bf16_wmma_kernel(
    const float* __restrict__ A, const float* __restrict__ Bsrc,
    void* __restrict__ Cout, const float* __restrict__ bias,
    const float* __restrict__ Res, int M, int N, int K,
    int relu, int outBf16)
{
  __shared__ __bf16 Atile[2][128 * 32];  // [m][k]
  __shared__ __bf16 Btile[2][128 * 32];  // transposed: [n][k]

  const int tid  = threadIdx.x;
  const int lane = tid & 31;
  const int w    = tid >> 5;
  const int wm   = w >> 1;                 // 0..3 -> rows wm*32
  const int wn   = w & 1;                  // 0..1 -> cols wn*64
  const int Nbase = blockIdx.x * 128;
  const int Mbase = blockIdx.y * 128;

  const int arow = tid >> 1, acol = (tid & 1) * 16;   // A staging: 16 f32/thread
  const int brow = tid >> 1, bcol = (tid & 1) * 16;   // B staging: 16 f32/thread

  float ra[16], rb[16];

  auto loadAB = [&](int kb) {
    const float* asrc = A + (size_t)(Mbase + arow) * K + kb + acol;
#pragma unroll
    for (int u = 0; u < 4; ++u) {
      float4 t = ((const float4*)asrc)[u];
      ra[4*u] = t.x; ra[4*u+1] = t.y; ra[4*u+2] = t.z; ra[4*u+3] = t.w;
    }
    if (TRANSB) {
      const float* bsrc = Bsrc + (size_t)(Nbase + brow) * K + kb + bcol;
#pragma unroll
      for (int u = 0; u < 4; ++u) {
        float4 t = ((const float4*)bsrc)[u];
        rb[4*u] = t.x; rb[4*u+1] = t.y; rb[4*u+2] = t.z; rb[4*u+3] = t.w;
      }
    } else {
#pragma unroll
      for (int j = 0; j < 16; ++j)
        rb[j] = Bsrc[(size_t)(kb + bcol + j) * N + Nbase + brow];
    }
  };
  auto storeAB = [&](int p) {
    __bf16* ad = &Atile[p][arow * 32 + acol];
    *(v8bf*)ad = pack8(ra); *(v8bf*)(ad + 8) = pack8(ra + 8);
    __bf16* bd = &Btile[p][brow * 32 + bcol];
    *(v8bf*)bd = pack8(rb); *(v8bf*)(bd + 8) = pack8(rb + 8);
  };

  v8f acc[2][4];
#pragma unroll
  for (int i = 0; i < 2; ++i)
#pragma unroll
    for (int j = 0; j < 4; ++j)
#pragma unroll
      for (int v = 0; v < 8; ++v) acc[i][j][v] = 0.0f;

  // pipeline prologue
  loadAB(0);
  storeAB(0);
  __syncthreads();

  const int nsteps = K / 32;
  for (int s = 0; s < nsteps; ++s) {
    const int p = s & 1;
    const bool more = (s + 1) < nsteps;
    if (more) loadAB((s + 1) * 32);   // global loads overlap the WMMAs below

    v16bf a0 = frag_a_lds(&Atile[p][(wm * 32 +  0) * 32], 32, lane);
    v16bf a1 = frag_a_lds(&Atile[p][(wm * 32 + 16) * 32], 32, lane);
#pragma unroll
    for (int j = 0; j < 4; ++j) {
      v16bf bf = frag_b_lds(&Btile[p][(wn * 64 + j * 16) * 32], 32, lane);
      acc[0][j] = __builtin_amdgcn_wmma_f32_16x16x32_bf16(false, a0, false, bf, (short)0, acc[0][j], false, false);
      acc[1][j] = __builtin_amdgcn_wmma_f32_16x16x32_bf16(false, a1, false, bf, (short)0, acc[1][j], false, false);
    }

    if (more) storeAB(p ^ 1);
    __syncthreads();
  }

  // epilogue: C-layout lane L: N = L&15, rows = v + 8*(L>>4)
  const int half = lane >> 4, nlo = lane & 15;
  float* Cf = (float*)Cout;
  __bf16* Cb = (__bf16*)Cout;
#pragma unroll
  for (int i = 0; i < 2; ++i) {
#pragma unroll
    for (int j = 0; j < 4; ++j) {
      const int col = Nbase + wn * 64 + j * 16 + nlo;
      const float bv = bias ? bias[col] : 0.0f;
#pragma unroll
      for (int v = 0; v < 8; ++v) {
        const int row = Mbase + wm * 32 + i * 16 + v + half * 8;
        float val = acc[i][j][v] + bv;
        if (relu) val = fmaxf(val, 0.0f);
        if (Res)  val += Res[(size_t)row * N + col];
        const size_t idx = (size_t)row * N + col;
        if (outBf16) Cb[idx] = (__bf16)val;
        else         Cf[idx] = val;
      }
    }
  }
}

// ---------------------------------------------------------------------------
// Flash attention (non-causal), bf16 Qh/Kh/Vh in memory, f32 out.
// Per block: 128 query rows of one (b,h); 8 waves x 16 rows; 32-key steps.
// K tile staged via CDNA5 async copy (global -> LDS, ASYNCcnt); softmax row
// reductions via v_permlane16 butterflies (no LDS round-trips).
// ---------------------------------------------------------------------------
__global__ __launch_bounds__(256) void mha_flash_wmma_kernel(
    const __bf16* __restrict__ Qh, const __bf16* __restrict__ Kh,
    const __bf16* __restrict__ Vh, float* __restrict__ Vatt)
{
  __shared__ __bf16 Klds[32 * 64];     // [key][dk]   (= B^T for scores)
  __shared__ __bf16 Vtl [64 * 32];     // [dv][key]   (= B^T for PV)
  __shared__ __bf16 Plds[8 * 16 * 32]; // per-wave P tile [16 rows][32 keys]

  const int tid  = threadIdx.x, lane = tid & 31, w = tid >> 5;
  const int half = lane >> 4, nlo = lane & 15;
  const int qblk = blockIdx.x * 128;
  const int h    = blockIdx.y;
  const int b    = blockIdx.z;
  const float scale = 1.0f / (8.0f + 1e-6f);   // 1/(sqrt(dk)+eps)

  // Q fragments (A-layout) straight from global bf16
  const int q = qblk + w * 16 + nlo;
  const __bf16* qrow = Qh + (((size_t)b * SEQ + q) * HEADS + h) * HDIM;
  v16bf qf[2];
#pragma unroll
  for (int c = 0; c < 2; ++c) {
    v8bf p0 = *(const v8bf*)(qrow + c * 32 + half * 8);
    v8bf p1 = *(const v8bf*)(qrow + c * 32 + 16 + half * 8);
    qf[c] = cat16(p0, p1);
  }

  float mrow[8], lrow[8];
  v8f o[4];
#pragma unroll
  for (int v = 0; v < 8; ++v) { mrow[v] = -3.0e38f; lrow[v] = 0.0f; }
#pragma unroll
  for (int nb = 0; nb < 4; ++nb)
#pragma unroll
    for (int v = 0; v < 8; ++v) o[nb][v] = 0.0f;

  for (int kb = 0; kb < SEQ; kb += 32) {
    const int key = tid >> 3, seg = (tid & 7) * 8;   // 8 bf16 = 16B per thread
    // stage K block [32][64] via async global->LDS b128 copy (no conversion)
    {
      const __bf16* gsrc = Kh + (((size_t)b * SEQ + kb + key) * HEADS + h) * HDIM + seg;
      unsigned long long ga = (unsigned long long)gsrc;
      unsigned int ldst = (unsigned int)(unsigned long long)(Klds + key * 64 + seg);
      asm volatile("global_load_async_to_lds_b128 %0, %1, off"
                   :: "v"(ldst), "v"(ga) : "memory");
    }
    // stage V block transposed -> Vtl[dv][key] (needs scatter, regular path)
    {
      v8bf t = *(const v8bf*)(Vh + (((size_t)b * SEQ + kb + key) * HEADS + h) * HDIM + seg);
#pragma unroll
      for (int j = 0; j < 8; ++j) Vtl[(seg + j) * 32 + key] = t[j];
    }
    asm volatile("s_wait_asynccnt 0x0" ::: "memory");
    __syncthreads();

    // scores: two 16x16 tiles (key groups), accumulate dk in 2 x32 steps
    v8f s[2];
#pragma unroll
    for (int kg = 0; kg < 2; ++kg) {
#pragma unroll
      for (int v = 0; v < 8; ++v) s[kg][v] = 0.0f;
#pragma unroll
      for (int c = 0; c < 2; ++c) {
        v16bf kf = frag_b_lds(Klds + (kg * 16) * 64 + c * 32, 64, lane);
        s[kg] = __builtin_amdgcn_wmma_f32_16x16x32_bf16(
            false, qf[c], false, kf, (short)0, s[kg], false, false);
      }
#pragma unroll
      for (int v = 0; v < 8; ++v) s[kg][v] *= scale;
    }

    // online softmax: rows live across 16-lane halves -> permlane butterflies
    float mnew[8], alpha[8];
#pragma unroll
    for (int v = 0; v < 8; ++v) {
      float mx = red16_max(fmaxf(s[0][v], s[1][v]));
      mnew[v]  = fmaxf(mrow[v], mx);
      alpha[v] = __expf(mrow[v] - mnew[v]);
      mrow[v]  = mnew[v];
    }
#pragma unroll
    for (int kg = 0; kg < 2; ++kg)
#pragma unroll
      for (int v = 0; v < 8; ++v)
        s[kg][v] = __expf(s[kg][v] - mnew[v]);
#pragma unroll
    for (int v = 0; v < 8; ++v) {
      float rs = red16_sum(s[0][v] + s[1][v]);
      lrow[v] = alpha[v] * lrow[v] + rs;
    }
#pragma unroll
    for (int nb = 0; nb < 4; ++nb)
#pragma unroll
      for (int v = 0; v < 8; ++v) o[nb][v] *= alpha[v];

    // reshape P: C-layout regs -> row-major LDS [16][32], read back as A-frag
    __bf16* P = Plds + w * (16 * 32);
#pragma unroll
    for (int kg = 0; kg < 2; ++kg)
#pragma unroll
      for (int v = 0; v < 8; ++v)
        P[(v + half * 8) * 32 + kg * 16 + nlo] = (__bf16)s[kg][v];
    asm volatile("s_wait_dscnt 0x0" ::: "memory");  // wave-internal LDS RAW

    v16bf pf = frag_a_lds(P, 32, lane);
#pragma unroll
    for (int nb = 0; nb < 4; ++nb) {
      v16bf vf = frag_b_lds(Vtl + (nb * 16) * 32, 32, lane);
      o[nb] = __builtin_amdgcn_wmma_f32_16x16x32_bf16(
          false, pf, false, vf, (short)0, o[nb], false, false);
    }
    __syncthreads();
  }

  // normalize and store f32: Vatt[b][q][h*64 + dv]
#pragma unroll
  for (int v = 0; v < 8; ++v) {
    const int row = qblk + w * 16 + v + half * 8;
    const float inv = 1.0f / lrow[v];
    float* outp = Vatt + ((size_t)b * SEQ + row) * (HEADS * HDIM) + h * HDIM;
#pragma unroll
    for (int nb = 0; nb < 4; ++nb)
      outp[nb * 16 + nlo] = o[nb][v] * inv;
  }
}

// ---------------------------------------------------------------------------
extern "C" void kernel_launch(void* const* d_in, const int* in_sizes, int n_in,
                              void* d_out, int out_size, void* d_ws, size_t ws_size,
                              hipStream_t stream) {
  const float* Q  = (const float*)d_in[0];
  const float* K  = (const float*)d_in[1];
  const float* V  = (const float*)d_in[2];
  const float* Wq = (const float*)d_in[3];
  const float* Wk = (const float*)d_in[4];
  const float* Wv = (const float*)d_in[5];
  const float* Wo = (const float*)d_in[6];
  const float* W1 = (const float*)d_in[7];
  const float* b1 = (const float*)d_in[8];
  const float* W2 = (const float*)d_in[9];
  const float* b2 = (const float*)d_in[10];
  float* out = (float*)d_out;

  const int M = MTOT, N = DIM, Kd = DIM;
  const size_t bufE = (size_t)M * DIM;

  // workspace: 3 bf16 buffers (Qh/Kh/Vh) + 3 f32 buffers (Vatt/X/H1)
  __bf16* Qh  = (__bf16*)d_ws;
  __bf16* Kh  = Qh + bufE;
  __bf16* Vh  = Kh + bufE;
  float* Vatt = (float*)(Vh + bufE);
  float* X    = Vatt + bufE;
  float* H1   = X + bufE;

  dim3 tb(256);
  dim3 gg(N / 128, M / 128);
  dim3 ga(SEQ / 128, HEADS, BATCH);

  // Q/K/V projections, bf16 outputs for the attention stage
  gemm_bf16_wmma_kernel<0><<<gg, tb, 0, stream>>>(Q, Wq, Qh, nullptr, nullptr, M, N, Kd, 0, 1);
  gemm_bf16_wmma_kernel<0><<<gg, tb, 0, stream>>>(K, Wk, Kh, nullptr, nullptr, M, N, Kd, 0, 1);
  gemm_bf16_wmma_kernel<0><<<gg, tb, 0, stream>>>(V, Wv, Vh, nullptr, nullptr, M, N, Kd, 0, 1);
  // attention
  mha_flash_wmma_kernel<<<ga, tb, 0, stream>>>(Qh, Kh, Vh, Vatt);
  // output projection + residual: X = Vatt @ Wo + Q
  gemm_bf16_wmma_kernel<0><<<gg, tb, 0, stream>>>(Vatt, Wo, X, nullptr, Q, M, N, Kd, 0, 0);
  // FFN1: H1 = relu(X @ W1.T + b1)
  gemm_bf16_wmma_kernel<1><<<gg, tb, 0, stream>>>(X, W1, H1, b1, nullptr, M, N, Kd, 1, 0);
  // FFN2 + residual: out = H1 @ W2.T + b2 + X
  gemm_bf16_wmma_kernel<1><<<gg, tb, 0, stream>>>(H1, W2, out, b2, X, M, N, Kd, 0, 0);
}
